// Controlled_NSDE_60361470378076
// MI455X (gfx1250) — matrix operand order repaired and checked by reading, not compiled, not run on hardware
//
#include <hip/hip_runtime.h>
#include <hip/hip_bf16.h>
#include <stdint.h>
#include <stddef.h>

// ---------------------------------------------------------------------------
// Controlled NSDE on MI455X (gfx1250).
// Strategy: batch rows are independent across the 199-step scan, so each
// block owns 64 batch rows and runs the whole time loop with state in LDS.
// All matmuls use v_wmma_f32_16x16x32_bf16 (8x K-throughput vs f32 WMMA);
// the SDE state update stays in fp32. Weights are pre-packed to bf16 in the
// exact WMMA B-fragment lane layout so the inner loop is pure b128 loads +
// WMMA. The t-column of alpha layer-1 is folded into a per-step bias table.
// ---------------------------------------------------------------------------

typedef __bf16 bf16_t;
typedef __attribute__((ext_vector_type(16))) __bf16 v16bf;
typedef __attribute__((ext_vector_type(8)))  __bf16 v8bf;
typedef __attribute__((ext_vector_type(8)))  float  v8f;

#define BATCH   4096
#define LSTEPS  200
#define DDIM    64
#define MROWS   64          // batch rows per block
#define MTILES  4           // 64 / 16
#define NWAVES  16
#define BLOCKSZ 512

// LDS strides (in elements) with padding to keep 16B alignment & spread banks
#define XF_STRIDE 68        // f32, 64x68
#define XA_STRIDE 136       // bf16, 64x136  ([x | a] concat buffer)
#define H_STRIDE  264       // bf16, 64x264 hidden activations

// dynamic LDS layout (bytes)
#define SM_XF 0             // 64*68*4  = 17408
#define SM_XA 17408         // 64*136*2 = 17408
#define SM_HA 34816         // 64*264*2 = 33792
#define SM_HB 68608         // 33792
#define SM_MU 102400        // 17408
#define SM_SG 119808        // 17408
#define SM_TOT 137216

// packed-weight offsets in workspace (bf16 elements), after 256KB f32 region
#define OFF_AW1 0           // 64x256
#define OFF_AW2 16384       // 256x256
#define OFF_AW3 81920       // 256x64
#define OFF_DW1 98304       // 128x256
#define OFF_DW2 131072
#define OFF_DW3 196608
#define OFF_SW1 212992      // 64x256
#define OFF_SW2 229376
#define OFF_SW3 294912
#define OFF_RW1 311296      // 128x256
#define OFF_RW2 344064
#define OFF_FW1 409600      // 64x256
#define OFF_FW2 425984      // ends 491520

// ---------------------------------------------------------------------------
// Prelude: pack a KxN fp32 weight (row-major, x @ W convention) into bf16
// WMMA B-fragments. Fragment f = kt*NT + nt holds a 32x16 tile; within it,
// lane l covers column n = nt*16 + (l&15), K-half = 16*(l>>4), and the 16
// bf16 elements per lane are consecutive K values (same K order as the A loads).
// ---------------------------------------------------------------------------
__global__ void nsde_pack_w(const float* __restrict__ W, bf16_t* __restrict__ dst,
                            int K, int N, int rowOff) {
    int idx = blockIdx.x * 256 + threadIdx.x;
    int total = K * N;
    if (idx >= total) return;
    int e    = idx & 15;
    int lane = (idx >> 4) & 31;
    int f    = idx >> 9;
    int NT   = N >> 4;
    int kt   = f / NT;
    int nt   = f - kt * NT;
    int k    = kt * 32 + ((lane >> 4) << 4) + e;
    int n    = nt * 16 + (lane & 15);
    dst[idx] = (bf16_t)W[(size_t)(k + rowOff) * N + n];
}

// alpha layer-1 effective bias per step: b1[n] + ts[s] * W1[0][n]
__global__ void nsde_alpha_bias(const float* __restrict__ ts,
                                const float* __restrict__ W1row0,
                                const float* __restrict__ b1,
                                float* __restrict__ tab) {
    int s = blockIdx.x;
    int n = threadIdx.x;
    tab[s * 256 + n] = b1[n] + ts[s] * W1row0[n];
}

// brownian_increments pass-through copy (float4 vectorized)
__global__ void nsde_copy4(const float4* __restrict__ src, float4* __restrict__ dst, int n) {
    int i = blockIdx.x * 256 + threadIdx.x;
    if (i < n) dst[i] = src[i];
}

// ---------------------------------------------------------------------------
// One FFN layer tile-GEMM on a 64-row activation block.
// MODE 0: dest = LDS bf16 (hidden layer), MODE 1: dest = LDS f32 (mu/sig),
// MODE 2: dest = xa[:,64:128] bf16 + global f32 (actions).
// ---------------------------------------------------------------------------
template <int KT, int NT, int MODE, bool RELU>
__device__ __forceinline__ void ffn_layer(
    const bf16_t* __restrict__ Bp,          // packed B fragments (global, bf16)
    const bf16_t* __restrict__ A, int aStride,   // LDS activations, bf16 row-major
    const float*  __restrict__ bias,        // per-column bias (>= NT*16 floats)
    bf16_t* __restrict__ Hb, int hStride,   // MODE 0/2 dest
    float*  __restrict__ Hf, int fStride,   // MODE 1 dest
    float*  __restrict__ Gout, int gRowStride) { // MODE 2 global dest
    static_assert(NWAVES % NT == 0, "NT must divide wave count");
    constexpr int WPC = NWAVES / NT;        // waves sharing a column stripe
    constexpr int MTC = MTILES / WPC;       // M-tiles per wave
    const int tid  = threadIdx.x;
    const int lane = tid & 31;
    const int wv   = tid >> 5;
    const int g    = lane >> 4;             // K-half selector
    const int ln16 = lane & 15;
    const int nt   = wv % NT;
    const int mt0  = (wv / NT) * MTC;

    const v8f vzero = {0.f, 0.f, 0.f, 0.f, 0.f, 0.f, 0.f, 0.f};
    v8f acc[MTC];
#pragma unroll
    for (int i = 0; i < MTC; ++i) acc[i] = vzero;

#pragma unroll
    for (int kt = 0; kt < KT; ++kt) {
        const bf16_t* bptr = Bp + (((size_t)(kt * NT + nt) * 32 + lane) * 16);
        union { v16bf v; v8bf h[2]; } bf;
        bf.h[0] = *reinterpret_cast<const v8bf*>(bptr);
        bf.h[1] = *reinterpret_cast<const v8bf*>(bptr + 8);
        if (kt + 1 < KT) {
            // next B fragment -> global_prefetch_b8 (L2 resident weights)
            __builtin_prefetch(Bp + (((size_t)((kt + 1) * NT + nt) * 32 + lane) * 16), 0, 0);
        }
#pragma unroll
        for (int i = 0; i < MTC; ++i) {
            const int m0 = (mt0 + i) * 16 + ln16;
            const bf16_t* arow = A + (size_t)m0 * aStride + kt * 32;
            union { v16bf v; v8bf h[2]; } af;
            af.h[0] = *reinterpret_cast<const v8bf*>(arow + 16 * g);      // K = 16g..16g+7
            af.h[1] = *reinterpret_cast<const v8bf*>(arow + 16 * g + 8);  // K = 16g+8..16g+15
            acc[i] = __builtin_amdgcn_wmma_f32_16x16x32_bf16(
                false, af.v, false, bf.v, (short)0, acc[i], false, false);
        }
    }

    const float bn = bias[nt * 16 + ln16];
#pragma unroll
    for (int i = 0; i < MTC; ++i) {
#pragma unroll
        for (int r = 0; r < 8; ++r) {
            float v = acc[i][r] + bn;
            if (RELU) v = v > 0.f ? v : 0.f;
            const int row = (mt0 + i) * 16 + 8 * g + r;   // D layout: M = r + 8*(lane/16)
            const int col = nt * 16 + ln16;               //           N = lane % 16
            if (MODE == 0) {
                Hb[(size_t)row * hStride + col] = (bf16_t)v;
            } else if (MODE == 1) {
                Hf[(size_t)row * fStride + col] = v;
            } else {
                Hb[(size_t)row * hStride + 64 + col] = (bf16_t)v;   // xa a-part
                Gout[(size_t)row * gRowStride + col] = v;           // actions
            }
        }
    }
}

struct NsdeParams {
    const float* ts;
    const float* x0;
    const float* dW;
    const bf16_t *aW1, *aW2, *aW3;
    const bf16_t *dW1, *dW2, *dW3;
    const bf16_t *sW1, *sW2, *sW3;
    const bf16_t *rW1, *rW2;
    const bf16_t *fW1, *fW2;
    const float* aBiasTab;                       // 199 x 256
    const float *ab2, *ab3;
    const float *db1, *db2, *db3;
    const float *sb1, *sb2, *sb3;
    const float *rb1, *rb2, *rW3, *rb3;
    const float *fb1, *fb2, *fW3, *fb3;
    float *out_x, *out_act, *out_rew;
};

__global__ __launch_bounds__(BLOCKSZ) void nsde_main(NsdeParams p) {
    extern __shared__ char smem[];
    float*  xF  = reinterpret_cast<float*>(smem + SM_XF);
    bf16_t* xa  = reinterpret_cast<bf16_t*>(smem + SM_XA);
    bf16_t* HA  = reinterpret_cast<bf16_t*>(smem + SM_HA);
    bf16_t* HB  = reinterpret_cast<bf16_t*>(smem + SM_HB);
    float*  muF = reinterpret_cast<float*>(smem + SM_MU);
    float*  sgF = reinterpret_cast<float*>(smem + SM_SG);

    const int tid = threadIdx.x;
    const int b0  = blockIdx.x * MROWS;

    // init state x = x0; emit x[:,0,:]
    for (int i = tid; i < MROWS * DDIM; i += BLOCKSZ) {
        int m = i >> 6, d = i & 63;
        float v = p.x0[(size_t)(b0 + m) * DDIM + d];
        xF[m * XF_STRIDE + d] = v;
        xa[m * XA_STRIDE + d] = (bf16_t)v;
        p.out_x[((size_t)(b0 + m) * LSTEPS) * DDIM + d] = v;
    }
    __syncthreads();

    for (int s = 0; s < LSTEPS - 1; ++s) {
        const float dt = p.ts[s + 1] - p.ts[s];

        // ---- alpha: tx = [t, x]  (t folded into per-step bias) ----
        ffn_layer<2, 16, 0, true >(p.aW1, xa, XA_STRIDE, p.aBiasTab + s * 256,
                                   HA, H_STRIDE, nullptr, 0, nullptr, 0);
        __syncthreads();
        ffn_layer<8, 16, 0, true >(p.aW2, HA, H_STRIDE, p.ab2,
                                   HB, H_STRIDE, nullptr, 0, nullptr, 0);
        __syncthreads();
        ffn_layer<8, 4, 2, false>(p.aW3, HB, H_STRIDE, p.ab3,
                                  xa, XA_STRIDE, nullptr, 0,
                                  p.out_act + ((size_t)b0 * (LSTEPS - 1) + s) * DDIM,
                                  (LSTEPS - 1) * DDIM);
        __syncthreads();

        // ---- drift: xa = [x, a], K = 128 ----
        ffn_layer<4, 16, 0, true >(p.dW1, xa, XA_STRIDE, p.db1,
                                   HA, H_STRIDE, nullptr, 0, nullptr, 0);
        __syncthreads();
        ffn_layer<8, 16, 0, true >(p.dW2, HA, H_STRIDE, p.db2,
                                   HB, H_STRIDE, nullptr, 0, nullptr, 0);
        __syncthreads();
        ffn_layer<8, 4, 1, false>(p.dW3, HB, H_STRIDE, p.db3,
                                  nullptr, 0, muF, XF_STRIDE, nullptr, 0);
        __syncthreads();

        // ---- running reward: xa, K = 128 ----
        ffn_layer<4, 16, 0, true >(p.rW1, xa, XA_STRIDE, p.rb1,
                                   HA, H_STRIDE, nullptr, 0, nullptr, 0);
        __syncthreads();
        ffn_layer<8, 16, 0, true >(p.rW2, HA, H_STRIDE, p.rb2,
                                   HB, H_STRIDE, nullptr, 0, nullptr, 0);
        __syncthreads();
        // 256->1 GEMV on waves 0-1, overlapped with diffusion layer 1 below
        if (tid < MROWS) {
            float sum = p.rb3[0];
            const bf16_t* hrow = HB + tid * H_STRIDE;
            for (int k = 0; k < 256; ++k) sum += (float)hrow[k] * p.rW3[k];
            p.out_rew[(size_t)(b0 + tid) * LSTEPS + s] = sum;
        }

        // ---- diffusion: x only, K = 64 ----
        ffn_layer<2, 16, 0, true >(p.sW1, xa, XA_STRIDE, p.sb1,
                                   HA, H_STRIDE, nullptr, 0, nullptr, 0);
        __syncthreads();
        ffn_layer<8, 16, 0, true >(p.sW2, HA, H_STRIDE, p.sb2,
                                   HB, H_STRIDE, nullptr, 0, nullptr, 0);
        __syncthreads();
        ffn_layer<8, 4, 1, false>(p.sW3, HB, H_STRIDE, p.sb3,
                                  nullptr, 0, sgF, XF_STRIDE, nullptr, 0);
        __syncthreads();

        // ---- fp32 state update: x += mu*dt + sig*dW ----
        for (int i = tid; i < MROWS * DDIM; i += BLOCKSZ) {
            int m = i >> 6, d = i & 63;
            float xv = xF[m * XF_STRIDE + d];
            float dw = p.dW[((size_t)(b0 + m) * (LSTEPS - 1) + s) * DDIM + d];
            float xn = xv + muF[m * XF_STRIDE + d] * dt + sgF[m * XF_STRIDE + d] * dw;
            xF[m * XF_STRIDE + d] = xn;
            xa[m * XA_STRIDE + d] = (bf16_t)xn;
            p.out_x[((size_t)(b0 + m) * LSTEPS + s + 1) * DDIM + d] = xn;
        }
        __syncthreads();
    }

    // ---- final reward FFN on x_last ----
    ffn_layer<2, 16, 0, true>(p.fW1, xa, XA_STRIDE, p.fb1,
                              HA, H_STRIDE, nullptr, 0, nullptr, 0);
    __syncthreads();
    ffn_layer<8, 16, 0, true>(p.fW2, HA, H_STRIDE, p.fb2,
                              HB, H_STRIDE, nullptr, 0, nullptr, 0);
    __syncthreads();
    if (tid < MROWS) {
        float sum = p.fb3[0];
        const bf16_t* hrow = HB + tid * H_STRIDE;
        for (int k = 0; k < 256; ++k) sum += (float)hrow[k] * p.fW3[k];
        p.out_rew[(size_t)(b0 + tid) * LSTEPS + (LSTEPS - 1)] = sum;
    }
}

// ---------------------------------------------------------------------------
extern "C" void kernel_launch(void* const* d_in, const int* in_sizes, int n_in,
                              void* d_out, int out_size, void* d_ws, size_t ws_size,
                              hipStream_t stream) {
    (void)in_sizes; (void)n_in; (void)out_size; (void)ws_size;
    const float* ts  = (const float*)d_in[0];
    const float* x0  = (const float*)d_in[1];
    const float* bw  = (const float*)d_in[2];
    const float* aW1f = (const float*)d_in[3];  const float* ab1 = (const float*)d_in[4];
    const float* aW2f = (const float*)d_in[5];  const float* ab2 = (const float*)d_in[6];
    const float* aW3f = (const float*)d_in[7];  const float* ab3 = (const float*)d_in[8];
    const float* dW1f = (const float*)d_in[9];  const float* db1 = (const float*)d_in[10];
    const float* dW2f = (const float*)d_in[11]; const float* db2 = (const float*)d_in[12];
    const float* dW3f = (const float*)d_in[13]; const float* db3 = (const float*)d_in[14];
    const float* sW1f = (const float*)d_in[15]; const float* sb1 = (const float*)d_in[16];
    const float* sW2f = (const float*)d_in[17]; const float* sb2 = (const float*)d_in[18];
    const float* sW3f = (const float*)d_in[19]; const float* sb3 = (const float*)d_in[20];
    const float* rW1f = (const float*)d_in[21]; const float* rb1 = (const float*)d_in[22];
    const float* rW2f = (const float*)d_in[23]; const float* rb2 = (const float*)d_in[24];
    const float* rW3f = (const float*)d_in[25]; const float* rb3 = (const float*)d_in[26];
    const float* fW1f = (const float*)d_in[27]; const float* fb1 = (const float*)d_in[28];
    const float* fW2f = (const float*)d_in[29]; const float* fb2 = (const float*)d_in[30];
    const float* fW3f = (const float*)d_in[31]; const float* fb3 = (const float*)d_in[32];

    float*  biasTab = (float*)d_ws;                              // 199*256 f32
    bf16_t* pk      = (bf16_t*)((char*)d_ws + 256 * 1024);       // packed bf16 weights

    auto packW = [&](const float* W, size_t off, int K, int N, int rowOff) {
        int total = K * N;
        nsde_pack_w<<<(total + 255) / 256, 256, 0, stream>>>(W, pk + off, K, N, rowOff);
    };
    packW(aW1f, OFF_AW1, 64, 256, 1);      // skip t-row (folded into bias)
    packW(aW2f, OFF_AW2, 256, 256, 0);
    packW(aW3f, OFF_AW3, 256, 64, 0);
    packW(dW1f, OFF_DW1, 128, 256, 0);
    packW(dW2f, OFF_DW2, 256, 256, 0);
    packW(dW3f, OFF_DW3, 256, 64, 0);
    packW(sW1f, OFF_SW1, 64, 256, 0);
    packW(sW2f, OFF_SW2, 256, 256, 0);
    packW(sW3f, OFF_SW3, 256, 64, 0);
    packW(rW1f, OFF_RW1, 128, 256, 0);
    packW(rW2f, OFF_RW2, 256, 256, 0);
    packW(fW1f, OFF_FW1, 64, 256, 0);
    packW(fW2f, OFF_FW2, 256, 256, 0);

    nsde_alpha_bias<<<LSTEPS - 1, 256, 0, stream>>>(ts, aW1f, ab1, biasTab);

    float* out_x   = (float*)d_out;
    float* out_bw  = out_x  + (size_t)BATCH * LSTEPS * DDIM;
    float* out_act = out_bw + (size_t)BATCH * (LSTEPS - 1) * DDIM;
    float* out_rew = out_act + (size_t)BATCH * (LSTEPS - 1) * DDIM;

    int n4 = BATCH * (LSTEPS - 1) * DDIM / 4;
    nsde_copy4<<<(n4 + 255) / 256, 256, 0, stream>>>((const float4*)bw, (float4*)out_bw, n4);

    NsdeParams p;
    p.ts = ts; p.x0 = x0; p.dW = bw;
    p.aW1 = pk + OFF_AW1; p.aW2 = pk + OFF_AW2; p.aW3 = pk + OFF_AW3;
    p.dW1 = pk + OFF_DW1; p.dW2 = pk + OFF_DW2; p.dW3 = pk + OFF_DW3;
    p.sW1 = pk + OFF_SW1; p.sW2 = pk + OFF_SW2; p.sW3 = pk + OFF_SW3;
    p.rW1 = pk + OFF_RW1; p.rW2 = pk + OFF_RW2;
    p.fW1 = pk + OFF_FW1; p.fW2 = pk + OFF_FW2;
    p.aBiasTab = biasTab;
    p.ab2 = ab2; p.ab3 = ab3;
    p.db1 = db1; p.db2 = db2; p.db3 = db3;
    p.sb1 = sb1; p.sb2 = sb2; p.sb3 = sb3;
    p.rb1 = rb1; p.rb2 = rb2; p.rW3 = rW3f; p.rb3 = rb3;
    p.fb1 = fb1; p.fb2 = fb2; p.fW3 = fW3f; p.fb3 = fb3;
    p.out_x = out_x; p.out_act = out_act; p.out_rew = out_rew;

    nsde_main<<<BATCH / MROWS, BLOCKSZ, SM_TOT, stream>>>(p);
}
